// MultiheadAttention_57561151701314
// MI455X (gfx1250) — compile-verified, matrix operand order checked by reading, not run on hardware
//
#include <hip/hip_runtime.h>
#include <hip/hip_bf16.h>

typedef __attribute__((ext_vector_type(16))) _Float16 v16h;
typedef __attribute__((ext_vector_type(8)))  _Float16 v8h;
typedef __attribute__((ext_vector_type(8)))  float    v8f;
typedef __attribute__((ext_vector_type(4)))  float    v4f;
typedef __attribute__((ext_vector_type(4)))  int      v4i;

#define NEG_INF (-__builtin_inff())

// Pack two contiguous groups of 8 halfs (at p and p+16) into a v16h WMMA fragment.
// Matches the 16-bit A/B VGPR layout: elems 0-7 = K base..base+7, elems 8-15 = K base+16..base+23.
__device__ __forceinline__ v16h pack16(const _Float16* p) {
    v8h lo = *(const v8h*)p;
    v8h hi = *(const v8h*)(p + 16);
    v16h r;
#pragma unroll
    for (int i = 0; i < 8; ++i) { r[i] = lo[i]; r[8 + i] = hi[i]; }
    return r;
}

__device__ __forceinline__ v8f wmma_f16(v16h a, v16h b, v8f c) {
    return __builtin_amdgcn_wmma_f32_16x16x32_f16(false, a, false, b, (short)0, c, false, false);
}

// ---------------------------------------------------------------------------
// Kernel 1: convert weights fp32 -> f16, stored N-major (transposed) so that
// WMMA B-fragments read 8 contiguous halfs along the contraction dim.
//   sec 0..2: WqT/WkT/WvT[n=h*64+j][d] = W[h][d][j]
//   sec 3   : WoT[e][d] = Wo[d][e]
// ---------------------------------------------------------------------------
__global__ void convert_weights(const float* __restrict__ wq,
                                const float* __restrict__ wk,
                                const float* __restrict__ wv,
                                const float* __restrict__ wo,
                                _Float16* __restrict__ w16) {
    int t = blockIdx.x * blockDim.x + threadIdx.x;   // 0 .. 4M-1
    int sec = t >> 20;
    int r   = t & ((1 << 20) - 1);
    int n   = r >> 10;        // transposed row (output col of original)
    int d   = r & 1023;       // contraction index
    float v;
    if (sec < 3) {
        const float* w = (sec == 0) ? wq : (sec == 1) ? wk : wv;
        int h = n >> 6, j = n & 63;
        v = w[(h << 16) + (d << 6) + j];   // h*1024*64 + d*64 + j
    } else {
        v = wo[d * 1024 + n];
    }
    w16[t] = (_Float16)v;
}

// ---------------------------------------------------------------------------
// Kernel 2: tiled GEMM  C[8192 x 1024] = A[8192 x 1024] * BT^T
//   BT is f16, N-major: BT[n][k].
//   Block 256 threads = 8 waves; block tile 128(M) x 64(N); wave tile 16 x 64.
//   MODE 0: store f16 row-major      MODE 1: store f16 V-transposed layout
//   MODE 2: store f32 row-major (final output)
// ---------------------------------------------------------------------------
template <typename AT, int MODE>
__global__ __launch_bounds__(256, 2)
void gemm_wmma(const AT* __restrict__ A, const _Float16* __restrict__ BT,
               void* __restrict__ Cout) {
    __shared__ _Float16 lds[64 * 32];   // B tile: [n(64)][k(32)]

    const int tid  = threadIdx.x;
    const int lane = tid & 31;
    const int wave = tid >> 5;
    const int half = lane >> 4;
    const int l16  = lane & 15;

    const int rowBase = blockIdx.x * 128 + wave * 16;
    const int colBase = blockIdx.y * 64;
    const int row     = rowBase + l16;

    v8f acc[4] = {};

    for (int kk = 0; kk < 1024; kk += 32) {
        // ---- cooperative stage of B tile (64n x 32k f16 = 4 KB) ----
        {
            int n  = tid >> 2;
            int kc = (tid & 3) << 3;
            *(v8h*)(&lds[n * 32 + kc]) =
                *(const v8h*)(BT + (size_t)(colBase + n) * 1024 + kk + kc);
        }
        __syncthreads();

        // ---- A fragment (direct from global, cvt f32->f16 if needed) ----
        v16h a;
        if constexpr (sizeof(AT) == 4) {
            const float* ap = (const float*)A + (size_t)row * 1024 + kk + half * 8;
            v4f f0 = *(const v4f*)(ap);
            v4f f1 = *(const v4f*)(ap + 4);
            v4f f2 = *(const v4f*)(ap + 16);
            v4f f3 = *(const v4f*)(ap + 20);
#pragma unroll
            for (int i = 0; i < 4; ++i) {
                a[i]      = (_Float16)f0[i];
                a[4 + i]  = (_Float16)f1[i];
                a[8 + i]  = (_Float16)f2[i];
                a[12 + i] = (_Float16)f3[i];
            }
        } else {
            const _Float16* ap = (const _Float16*)A + (size_t)row * 1024 + kk + half * 8;
            a = pack16(ap);
        }

        // ---- 4 WMMAs over the 64-wide N strip ----
#pragma unroll
        for (int nt = 0; nt < 4; ++nt) {
            const _Float16* bp = &lds[(nt * 16 + l16) * 32 + half * 8];
            v16h b = pack16(bp);
            acc[nt] = wmma_f16(a, b, acc[nt]);
        }
        __syncthreads();
    }

    // ---- store ----
#pragma unroll
    for (int nt = 0; nt < 4; ++nt) {
#pragma unroll
        for (int i = 0; i < 8; ++i) {
            int rowT = rowBase + i + half * 8;        // D layout: M = vgpr (+8 hi half)
            int cn   = colBase + nt * 16 + l16;       // N across lanes
            float v  = acc[nt][i];
            if constexpr (MODE == 0) {
                ((_Float16*)Cout)[(size_t)rowT * 1024 + cn] = (_Float16)v;
            } else if constexpr (MODE == 1) {
                // Vt[((b*16+h)*64 + j)][s]  with rowT = b*1024+s, cn = h*64+j
                int bb = rowT >> 10, s = rowT & 1023;
                int hh = cn >> 6,   j = cn & 63;
                ((_Float16*)Cout)[((size_t)((bb * 16 + hh) * 64 + j)) * 1024 + s] = (_Float16)v;
            } else {
                ((float*)Cout)[(size_t)rowT * 1024 + cn] = v;
            }
        }
    }
}

// ---------------------------------------------------------------------------
// Kernel 3: flash attention, one wave per (b, h, 16-query block).
//   Scores computed transposed: S^T[key][q] = K_tile x Q^T  -> softmax over
//   keys is lane-local (per-lane q = lane&15); key tiles processed in pairs
//   so the exp'd P packs directly into a K=32 A-fragment with no shuffles.
// ---------------------------------------------------------------------------
__global__ __launch_bounds__(32)
void attn_kernel(const _Float16* __restrict__ Q, const _Float16* __restrict__ K,
                 const _Float16* __restrict__ Vt, const int* __restrict__ mask,
                 _Float16* __restrict__ Cc) {
    const int lane = threadIdx.x & 31;
    const int half = lane >> 4;
    const int l16  = lane & 15;

    const int qt = blockIdx.x & 63;
    const int h  = (blockIdx.x >> 6) & 15;
    const int b  = (int)(blockIdx.x >> 10);

    // Q^T B-fragments (per-lane column q = l16), loaded once.
    const _Float16* qp =
        Q + ((size_t)(b * 1024 + qt * 16 + l16)) * 1024 + h * 64 + half * 8;
    v16h bq0 = pack16(qp);        // j = 0..31
    v16h bq1 = pack16(qp + 32);   // j = 32..63

    const int* mrow = mask + (size_t)b * 1024 * 1024 + (size_t)(qt * 16 + l16) * 1024;

    v8f  o[4] = {};
    float m_run = NEG_INF, l_run = 0.f;

    for (int mt = 0; mt < 64; mt += 2) {
        float p[2][8];
        float tmax = NEG_INF;

#pragma unroll
        for (int t = 0; t < 2; ++t) {
            int keyrow = (mt + t) * 16 + l16;
            const _Float16* kp =
                K + ((size_t)(b * 1024 + keyrow)) * 1024 + h * 64 + half * 8;
            v16h ak0 = pack16(kp);
            v16h ak1 = pack16(kp + 32);
            v8f s = {};
            s = wmma_f16(ak0, bq0, s);
            s = wmma_f16(ak1, bq1, s);

            const int* mp = mrow + (mt + t) * 16 + half * 8;
            v4i m0 = *(const v4i*)mp;
            v4i m1 = *(const v4i*)(mp + 4);
#pragma unroll
            for (int i = 0; i < 8; ++i) {
                int  mk = (i < 4) ? m0[i] : m1[i - 4];
                float sv = s[i] * 0.125f;          // 1/sqrt(64)
                sv = (mk == 0) ? NEG_INF : sv;
                p[t][i] = sv;
                tmax = fmaxf(tmax, sv);
            }
        }

        // merge the two lane halves' key subsets, online softmax update
        tmax = fmaxf(tmax, __shfl_xor(tmax, 16));
        float m_new = fmaxf(m_run, tmax);
        float scl   = (m_run == NEG_INF) ? 0.f : __expf(m_run - m_new);
        float msafe = (m_new == NEG_INF) ? 0.f : m_new;

        float lt = 0.f;
        v16h ap;
#pragma unroll
        for (int i = 0; i < 8; ++i) {
            float e0 = __expf(p[0][i] - msafe);
            float e1 = __expf(p[1][i] - msafe);
            lt += e0 + e1;
            ap[i]     = (_Float16)e0;   // pair-local K = half*8 + i
            ap[8 + i] = (_Float16)e1;   // pair-local K = 16 + half*8 + i
        }
        lt += __shfl_xor(lt, 16);
        l_run = l_run * scl + lt;

        // rescale O: output row q = i + half*8, scale held by lane q
#pragma unroll
        for (int i = 0; i < 8; ++i) {
            float f = __shfl(scl, i + half * 8);
#pragma unroll
            for (int jt = 0; jt < 4; ++jt) o[jt][i] *= f;
        }

        // O += P @ V  (K = 32 keys of this pair; V pre-transposed: Vt[j][s])
#pragma unroll
        for (int jt = 0; jt < 4; ++jt) {
            const _Float16* vp =
                Vt + ((size_t)((b * 16 + h) * 64 + jt * 16 + l16)) * 1024 +
                mt * 16 + half * 8;
            v16h bv = pack16(vp);
            o[jt] = wmma_f16(ap, bv, o[jt]);
        }

        m_run = m_new;
    }

    // normalize and store f16 head-concat: Cc[b*1024+q][h*64+j]
#pragma unroll
    for (int i = 0; i < 8; ++i) {
        float l   = __shfl(l_run, i + half * 8);
        float inv = (l > 0.f) ? 1.f / l : 0.f;
        int qrow  = qt * 16 + i + half * 8;
        size_t base = ((size_t)(b * 1024 + qrow)) * 1024 + h * 64;
#pragma unroll
        for (int jt = 0; jt < 4; ++jt)
            Cc[base + jt * 16 + l16] = (_Float16)(o[jt][i] * inv);
    }
}

// ---------------------------------------------------------------------------
// Host-side launch. Workspace layout (f16 elements):
//   [0,1M)  WqT  [1M,2M) WkT  [2M,3M) WvT  [3M,4M) WoT
//   [4M,12M) Q16  [12M,20M) K16  [20M,28M) Vt  [28M,36M) C16   (72 MB total)
// ---------------------------------------------------------------------------
extern "C" void kernel_launch(void* const* d_in, const int* in_sizes, int n_in,
                              void* d_out, int out_size, void* d_ws, size_t ws_size,
                              hipStream_t stream) {
    const float* q    = (const float*)d_in[0];
    const float* k    = (const float*)d_in[1];
    const float* v    = (const float*)d_in[2];
    const int*   msk  = (const int*)d_in[3];
    const float* wq   = (const float*)d_in[4];
    const float* wk   = (const float*)d_in[5];
    const float* wv   = (const float*)d_in[6];
    const float* wo   = (const float*)d_in[7];
    float*       out  = (float*)d_out;

    _Float16* ws  = (_Float16*)d_ws;
    const size_t M1 = (size_t)1 << 20;
    _Float16* WqT = ws;
    _Float16* WkT = ws + 1 * M1;
    _Float16* WvT = ws + 2 * M1;
    _Float16* WoT = ws + 3 * M1;
    _Float16* Q16 = ws + 4 * M1;
    _Float16* K16 = ws + 12 * M1;
    _Float16* Vt  = ws + 20 * M1;
    _Float16* C16 = ws + 28 * M1;

    convert_weights<<<16384, 256, 0, stream>>>(wq, wk, wv, wo, ws);

    dim3 g(64, 16);
    gemm_wmma<float, 0><<<g, 256, 0, stream>>>(q, WqT, (void*)Q16);
    gemm_wmma<float, 0><<<g, 256, 0, stream>>>(k, WkT, (void*)K16);
    gemm_wmma<float, 1><<<g, 256, 0, stream>>>(v, WvT, (void*)Vt);

    attn_kernel<<<8192, 32, 0, stream>>>(Q16, K16, Vt, msk, C16);

    gemm_wmma<_Float16, 2><<<g, 256, 0, stream>>>(C16, WoT, (void*)out);
}
